// SimpleStateSpaceLayer_14559939133721
// MI455X (gfx1250) — matrix operand order repaired and checked by reading, not compiled, not run on hardware
//
#include <hip/hip_runtime.h>

typedef __attribute__((ext_vector_type(16))) __bf16 v16bf;
typedef __attribute__((ext_vector_type(8)))  __bf16 v8bf;
typedef __attribute__((ext_vector_type(8)))  float  v8f;

constexpr int B_SZ = 4, SEQ = 4096, DM = 1024, DS = 16;
constexpr int NCHUNK = 32, CLEN = SEQ / NCHUNK;   // 32 chunks of 128 steps
constexpr float LN_EPS = 1e-5f;

// ---------------- one-time f32 -> bf16 conversion (8 elements / thread) ----------------
__global__ void __launch_bounds__(256)
f32_to_bf16_kernel(const float* __restrict__ src, __bf16* __restrict__ dst, int n8) {
  int i = blockIdx.x * 256 + threadIdx.x;
  if (i >= n8) return;
  const float4* s = reinterpret_cast<const float4*>(src) + 2 * (size_t)i;
  float4 a = s[0], b = s[1];
  v8bf v;
  v[0] = (__bf16)a.x; v[1] = (__bf16)a.y; v[2] = (__bf16)a.z; v[3] = (__bf16)a.w;
  v[4] = (__bf16)b.x; v[5] = (__bf16)b.y; v[6] = (__bf16)b.z; v[7] = (__bf16)b.w;
  *(reinterpret_cast<v8bf*>(dst) + i) = v;
}

// A fragment: 16 rows (m0..m0+15) x K (k0..k0+31) from row-major bf16 [., DM].
// lanes 0-15: M=lane, K = k0+{0..15}; lanes 16-31: M=lane-16, K = k0+{16..31}
static __device__ __forceinline__ v16bf load_afrag_bf(const __bf16* __restrict__ xb,
                                                      int m0, int k0, int lane) {
  int row  = m0 + (lane & 15);
  int half = lane >> 4;
  const __bf16* p = xb + (size_t)row * DM + k0 + half * 16;
  v8bf a0 = *reinterpret_cast<const v8bf*>(p);        // 16B load
  v8bf a1 = *reinterpret_cast<const v8bf*>(p + 8);    // 16B load
  return __builtin_shufflevector(a0, a1, 0, 1, 2, 3, 4, 5, 6, 7,
                                 8, 9, 10, 11, 12, 13, 14, 15);
}

// B fragment row: lane = N column holding 16 contiguous K values (32B, aligned)
static __device__ __forceinline__ v16bf load_brow_bf(const __bf16* __restrict__ p) {
  return *reinterpret_cast<const v16bf*>(p);
}

// ---------------- Bx / Cx projections: [16384,1024] x [1024,16] (x2) ----------------
__global__ void __launch_bounds__(256)
proj_bc_kernel(const __bf16* __restrict__ xb, const __bf16* __restrict__ Bw,
               const __bf16* __restrict__ Cw, float* __restrict__ Bx,
               float* __restrict__ Cx) {
  int lane = threadIdx.x & 31;
  int wave = threadIdx.x >> 5;
  int m0 = (blockIdx.x * 8 + wave) * 16;
  int half = lane >> 4;
  int n = lane & 15;
  v8f accB = {}; v8f accC = {};
  for (int k0 = 0; k0 < DM; k0 += 32) {
    v16bf A = load_afrag_bf(xb, m0, k0, lane);
    int kc = k0 + half * 16;
    v16bf Bf = load_brow_bf(Bw + (size_t)n * DM + kc);
    v16bf Cf = load_brow_bf(Cw + (size_t)n * DM + kc);
    accB = __builtin_amdgcn_wmma_f32_16x16x32_bf16(false, A, false, Bf,
                                                   (short)0, accB, false, false);
    accC = __builtin_amdgcn_wmma_f32_16x16x32_bf16(false, A, false, Cf,
                                                   (short)0, accC, false, false);
  }
#pragma unroll
  for (int i = 0; i < 8; ++i) {
    int r = i + half * 8;                 // C layout: vgpr i -> M=i (lanes<16) / M=8+i
    Bx[(m0 + r) * DS + n] = accB[i];
    Cx[(m0 + r) * DS + n] = accC[i];
  }
}

// ---------------- chunked scan, phase 1: local scan per chunk (h0 = 0) ----------------
__global__ void __launch_bounds__(256)
scan_local_kernel(const float* __restrict__ x, const float* __restrict__ A_log,
                  const float* __restrict__ Bx, float* __restrict__ hloc) {
  int idx = blockIdx.x * 256 + threadIdx.x;          // over b * NCHUNK * DM
  int d = idx % DM;
  int c = (idx / DM) % NCHUNK;
  int b = idx / (DM * NCHUNK);
  float A[DS], h[DS];
#pragma unroll
  for (int nn = 0; nn < DS; ++nn) { A[nn] = __expf(A_log[d * DS + nn]); h[nn] = 0.f; }
  int t0 = c * CLEN;
  for (int t = t0; t < t0 + CLEN; ++t) {
    float xv = x[(b * SEQ + t) * DM + d];
    const float* br = Bx + (b * SEQ + t) * DS;
#pragma unroll
    for (int nn = 0; nn < DS; ++nn) h[nn] = h[nn] * A[nn] + br[nn] * xv;
  }
  float* o = hloc + ((size_t)(c * B_SZ + b) * DM + d) * DS;
#pragma unroll
  for (int nn = 0; nn < DS; ++nn) o[nn] = h[nn];
}

// ---- phase 2: serial chunk prefix; converts hloc[c] into chunk-initial states in place
__global__ void __launch_bounds__(256)
scan_prefix_kernel(const float* __restrict__ A_log, float* __restrict__ hloc) {
  int idx = blockIdx.x * 256 + threadIdx.x;          // over b * DM
  int d = idx % DM;
  int b = idx / DM;
  float AL[DS], run[DS];
#pragma unroll
  for (int nn = 0; nn < DS; ++nn) {
    AL[nn] = __expf((float)CLEN * A_log[d * DS + nn]);   // A^CLEN in closed form
    run[nn] = 0.f;
  }
  for (int c = 0; c < NCHUNK; ++c) {
    float* p = hloc + ((size_t)(c * B_SZ + b) * DM + d) * DS;
#pragma unroll
    for (int nn = 0; nn < DS; ++nn) {
      float t = p[nn];
      p[nn] = run[nn];                    // exclusive prefix -> initial state of chunk c
      run[nn] = AL[nn] * run[nn] + t;
    }
  }
}

// ---- phase 3: replay with correct initial state; y = h.C + D*x, written to d_out ----
__global__ void __launch_bounds__(256)
scan_out_kernel(const float* __restrict__ x, const float* __restrict__ A_log,
                const float* __restrict__ Bx, const float* __restrict__ Cx,
                const float* __restrict__ Dv, const float* __restrict__ hloc,
                float* __restrict__ y) {
  int idx = blockIdx.x * 256 + threadIdx.x;
  int d = idx % DM;
  int c = (idx / DM) % NCHUNK;
  int b = idx / (DM * NCHUNK);
  float A[DS], h[DS];
  const float* hi = hloc + ((size_t)(c * B_SZ + b) * DM + d) * DS;
#pragma unroll
  for (int nn = 0; nn < DS; ++nn) { A[nn] = __expf(A_log[d * DS + nn]); h[nn] = hi[nn]; }
  float dv = Dv[d];
  int t0 = c * CLEN;
  for (int t = t0; t < t0 + CLEN; ++t) {
    float xv = x[(b * SEQ + t) * DM + d];
    const float* br = Bx + (b * SEQ + t) * DS;
    const float* cr = Cx + (b * SEQ + t) * DS;
    float acc = dv * xv;
#pragma unroll
    for (int nn = 0; nn < DS; ++nn) {
      h[nn] = h[nn] * A[nn] + br[nn] * xv;
      acc += h[nn] * cr[nn];
    }
    y[(b * SEQ + t) * DM + d] = acc;
  }
}

// ---- gate GEMM (WMMA bf16) + sigmoid gating + LayerNorm, in-place on d_out ----
// block = 1024 threads = 32 waves; block owns one 16-row M tile; wave w owns N tiles 2w, 2w+1
__global__ void __launch_bounds__(1024)
gate_ln_kernel(const float* __restrict__ x, const __bf16* __restrict__ xb,
               const __bf16* __restrict__ Gw, const float* __restrict__ Gb,
               const float* __restrict__ lng, const float* __restrict__ lnb,
               float* __restrict__ out) {
  __shared__ float s_sum[16];
  __shared__ float s_sq[16];
  int tid = threadIdx.x;
  int lane = tid & 31, wave = tid >> 5;
  int half = lane >> 4;
  int m0 = blockIdx.x * 16;
  if (tid < 16) { s_sum[tid] = 0.f; s_sq[tid] = 0.f; }
  __syncthreads();

  int col0 = (2 * wave) * 16 + (lane & 15);
  int col1 = col0 + 16;
  v8f acc0 = {}; v8f acc1 = {};
  for (int k0 = 0; k0 < DM; k0 += 32) {
    v16bf A = load_afrag_bf(xb, m0, k0, lane);
    int kc = k0 + half * 16;
    v16bf Bf0 = load_brow_bf(Gw + (size_t)col0 * DM + kc);
    v16bf Bf1 = load_brow_bf(Gw + (size_t)col1 * DM + kc);
    acc0 = __builtin_amdgcn_wmma_f32_16x16x32_bf16(false, A, false, Bf0,
                                                   (short)0, acc0, false, false);
    acc1 = __builtin_amdgcn_wmma_f32_16x16x32_bf16(false, A, false, Bf1,
                                                   (short)0, acc1, false, false);
  }

  float gb0 = Gb[col0], gb1 = Gb[col1];
  float val0[8], val1[8];
#pragma unroll
  for (int i = 0; i < 8; ++i) {
    int r = i + half * 8;
    int m = m0 + r;
    float g0 = 1.f / (1.f + __expf(-(acc0[i] + gb0)));
    float g1 = 1.f / (1.f + __expf(-(acc1[i] + gb1)));
    float y0 = out[(size_t)m * DM + col0], x0 = x[(size_t)m * DM + col0];
    float y1 = out[(size_t)m * DM + col1], x1 = x[(size_t)m * DM + col1];
    float v0 = g0 * y0 + (1.f - g0) * x0;
    float v1 = g1 * y1 + (1.f - g1) * x1;
    val0[i] = v0; val1[i] = v1;
    atomicAdd(&s_sum[r], v0 + v1);            // ds_add_f32; 1024 contributions per row
    atomicAdd(&s_sq[r], v0 * v0 + v1 * v1);
  }
  __syncthreads();

  float w0 = lng[col0], b0 = lnb[col0];
  float w1 = lng[col1], b1 = lnb[col1];
#pragma unroll
  for (int i = 0; i < 8; ++i) {
    int r = i + half * 8;
    int m = m0 + r;
    float mu = s_sum[r] * (1.f / DM);
    float var = s_sq[r] * (1.f / DM) - mu * mu;
    float rstd = rsqrtf(var + LN_EPS);
    out[(size_t)m * DM + col0] = (val0[i] - mu) * rstd * w0 + b0;
    out[(size_t)m * DM + col1] = (val1[i] - mu) * rstd * w1 + b1;
  }
}

extern "C" void kernel_launch(void* const* d_in, const int* in_sizes, int n_in,
                              void* d_out, int out_size, void* d_ws, size_t ws_size,
                              hipStream_t stream) {
  const float* x     = (const float*)d_in[0];
  const float* A_log = (const float*)d_in[1];
  const float* Bw    = (const float*)d_in[2];
  const float* Cw    = (const float*)d_in[3];
  const float* Dv    = (const float*)d_in[4];
  const float* Gw    = (const float*)d_in[5];
  const float* Gb    = (const float*)d_in[6];
  const float* lng   = (const float*)d_in[7];
  const float* lnb   = (const float*)d_in[8];
  float* out = (float*)d_out;

  // workspace carve-up (all offsets 32B-aligned)
  char* wsb = (char*)d_ws;
  size_t off = 0;
  __bf16* x_bf  = (__bf16*)(wsb + off); off += (size_t)B_SZ * SEQ * DM * 2;  // 32 MB
  __bf16* Gw_bf = (__bf16*)(wsb + off); off += (size_t)DM * DM * 2;          //  2 MB
  __bf16* Bw_bf = (__bf16*)(wsb + off); off += (size_t)DS * DM * 2;          // 32 KB
  __bf16* Cw_bf = (__bf16*)(wsb + off); off += (size_t)DS * DM * 2;          // 32 KB
  float*  Bx    = (float*)(wsb + off);  off += (size_t)B_SZ * SEQ * DS * 4;  //  1 MB
  float*  Cx    = (float*)(wsb + off);  off += (size_t)B_SZ * SEQ * DS * 4;  //  1 MB
  float*  hloc  = (float*)(wsb + off);  // NCHUNK*B*DM*DS*4 = 8 MB

  // one-time bf16 conversions
  int n8x = B_SZ * SEQ * DM / 8, n8g = DM * DM / 8, n8w = DS * DM / 8;
  f32_to_bf16_kernel<<<(n8x + 255) / 256, 256, 0, stream>>>(x,  x_bf,  n8x);
  f32_to_bf16_kernel<<<(n8g + 255) / 256, 256, 0, stream>>>(Gw, Gw_bf, n8g);
  f32_to_bf16_kernel<<<(n8w + 255) / 256, 256, 0, stream>>>(Bw, Bw_bf, n8w);
  f32_to_bf16_kernel<<<(n8w + 255) / 256, 256, 0, stream>>>(Cw, Cw_bf, n8w);

  proj_bc_kernel<<<(B_SZ * SEQ / 16) / 8, 256, 0, stream>>>(x_bf, Bw_bf, Cw_bf, Bx, Cx);
  scan_local_kernel<<<(B_SZ * NCHUNK * DM) / 256, 256, 0, stream>>>(x, A_log, Bx, hloc);
  scan_prefix_kernel<<<(B_SZ * DM) / 256, 256, 0, stream>>>(A_log, hloc);
  scan_out_kernel<<<(B_SZ * NCHUNK * DM) / 256, 256, 0, stream>>>(x, A_log, Bx, Cx, Dv,
                                                                  hloc, out);
  gate_ln_kernel<<<(B_SZ * SEQ) / 16, 1024, 0, stream>>>(x, x_bf, Gw_bf, Gb, lng, lnb, out);
}
